// SelfAttention_2027224564615
// MI455X (gfx1250) — matrix operand order repaired and checked by reading, not compiled, not run on hardware
//
#include <hip/hip_runtime.h>
#include <hip/hip_bf16.h>
#include <math.h>

typedef __attribute__((ext_vector_type(16))) _Float16 v16h;
typedef __attribute__((ext_vector_type(8)))  float    v8f;

#define DIM   1024
#define NHEAD 16
#define HDIM  64
#define BATCH 4
#define SEQ   2048
#define MTOT  (BATCH * SEQ)   // 8192
#define ATT_SCALE 0.125f      // 1/sqrt(64)

#define BM 128
#define BN 128
#define BK 32
#define LDP 40   // padded LDS row stride in f16 (80B: 16B-aligned, avoids bank conflicts)

#if __has_builtin(__builtin_amdgcn_tensor_load_to_lds)
#define USE_TDM 1
#else
#define USE_TDM 0
#endif

union Frag { v16h h; uint4 u[2]; };

// A fragment (16x32 f16, MxK). ISA layout: lanes 0-15 hold K=0..7 & 16..23,
// lanes 16-31 hold K=8..15 & 24..31, M = lane&15.
__device__ __forceinline__ v16h load_a_frag(const _Float16* base, int row0, int kcol0, int stride) {
  const int lane = threadIdx.x & 31;
  const int r  = row0 + (lane & 15);
  const int kk = kcol0 + ((lane >> 4) << 3);
  Frag f;
  f.u[0] = *(const uint4*)(base + (size_t)r * stride + kk);
  f.u[1] = *(const uint4*)(base + (size_t)r * stride + kk + 16);
  return f.h;
}

// B fragment (32x16 f16, KxN) from memory stored transposed as [N][K].
// ISA layout: lanes 0-15 hold K=0..7 & 16..23, lanes 16-31 hold K=8..15 & 24..31, N = lane&15.
__device__ __forceinline__ v16h load_b_frag(const _Float16* base, int ncol0, int kcol0, int stride) {
  const int lane = threadIdx.x & 31;
  const int n  = ncol0 + (lane & 15);
  const int kk = kcol0 + ((lane >> 4) << 3);
  Frag f;
  f.u[0] = *(const uint4*)(base + (size_t)n * stride + kk);
  f.u[1] = *(const uint4*)(base + (size_t)n * stride + kk + 16);
  return f.h;
}

__device__ __forceinline__ v8f wmma_f16(v16h a, v16h b, v8f c) {
  return __builtin_amdgcn_wmma_f32_16x16x32_f16(false, a, false, b, (short)0, c, false, false);
}

#if USE_TDM
typedef unsigned int v4u __attribute__((ext_vector_type(4)));
typedef int          v4i __attribute__((ext_vector_type(4)));
typedef int          v8i __attribute__((ext_vector_type(8)));

// LDS byte offset = low 32 bits of the aperture-based flat address (ISA 10.2).
__device__ __forceinline__ unsigned lds_off_of(const void* p) {
  return (unsigned)(size_t)p;
}

// TDM: DMA one [BM rows x BK f16] tile (row stride DIM elems) from global
// into LDS, padded to LDP f16 per row (pad 4 DWORDs after every 16 DWORDs).
__device__ __forceinline__ void tdm_load_tile_f16(const _Float16* gbase, unsigned lds_off,
                                                  unsigned tensorRows) {
  unsigned long long ga = (unsigned long long)(size_t)gbase;
  v4u g0;
  g0[0] = 1u;                                                  // count=1, user mode
  g0[1] = lds_off;                                             // lds_addr
  g0[2] = (unsigned)ga;                                        // global_addr[31:0]
  g0[3] = (unsigned)((ga >> 32) & 0x01FFFFFFu) | (2u << 30);   // global_addr[56:32] | type=2
  const unsigned dim0 = DIM;                                   // row length (elements)
  v8i g1;
  g1[0] = (int)((1u << 16)      // data_size = 2 bytes
              | (1u << 20)      // pad_enable
              | (3u << 22)      // pad_interval: 16 DWORDs (64B) between pads
              | (3u << 25));    // pad_amount: 4 DWORDs (16B) -> 80B row stride
  g1[1] = (int)((dim0 & 0xFFFFu) << 16);                        // tensor_dim0[15:0]
  g1[2] = (int)((dim0 >> 16) | ((tensorRows & 0xFFFFu) << 16)); // dim0 hi | tensor_dim1 lo
  g1[3] = (int)((tensorRows >> 16) | ((unsigned)BK << 16));     // dim1 hi | tile_dim0 = 32
  g1[4] = (int)BM;                                              // tile_dim1 = 128, tile_dim2 = 0
  g1[5] = (int)dim0;                                            // tensor_dim0_stride[31:0]
  g1[6] = 0;                                                    // stride hi | dim1_stride lo
  g1[7] = 0;
  v4i gz = {0, 0, 0, 0};
#if defined(__clang_major__) && (__clang_major__ >= 23)
  v8i gz8 = {0, 0, 0, 0, 0, 0, 0, 0};
  __builtin_amdgcn_tensor_load_to_lds(g0, g1, gz, gz, gz8, 0);
#else
  __builtin_amdgcn_tensor_load_to_lds(g0, g1, gz, gz, 0);
#endif
}
#endif // USE_TDM

// ---------------------------------------------------------------------------
// Conversions
// ---------------------------------------------------------------------------
__global__ void cvt_f32_f16(const float* __restrict__ in, _Float16* __restrict__ out, int n) {
  int i = blockIdx.x * blockDim.x + threadIdx.x;
  if (i < n) out[i] = (_Float16)in[i];
}

// w[K][N] (row-major) -> wt[N][K] f16
__global__ void cvt_transpose(const float* __restrict__ w, _Float16* __restrict__ wt, int K, int N) {
  int i = blockIdx.x * blockDim.x + threadIdx.x;
  if (i < K * N) {
    int k = i / N, n = i % N;
    wt[(size_t)n * K + k] = (_Float16)w[i];
  }
}

// ---------------------------------------------------------------------------
// Shared GEMM mainloop body (accumulates 2x4 WMMA tiles per wave).
// ---------------------------------------------------------------------------
__device__ __forceinline__ void gemm_tile_step(const _Float16* Acur, const _Float16* Bcur,
                                               int wm, int wn, v8f acc[2][4]) {
  v16h af[2], bf[4];
#pragma unroll
  for (int mi = 0; mi < 2; ++mi) af[mi] = load_a_frag(Acur, wm * 32 + mi * 16, 0, LDP);
#pragma unroll
  for (int ni = 0; ni < 4; ++ni) bf[ni] = load_b_frag(Bcur, wn * 64 + ni * 16, 0, LDP);
#pragma unroll
  for (int mi = 0; mi < 2; ++mi)
#pragma unroll
    for (int ni = 0; ni < 4; ++ni)
      acc[mi][ni] = wmma_f16(af[mi], bf[ni], acc[mi][ni]);
}

// ---------------------------------------------------------------------------
// GEMM1: qkv = x16 @ w_qkv (Wt pre-transposed [3072][1024]).
// Epilogue scatters into Q[B,H,T,d], K[B,H,T,d], V^T[B,H,d,T], all f16.
// ---------------------------------------------------------------------------
__global__ __launch_bounds__(256) void gemm_qkv(const _Float16* __restrict__ X,
                                                const _Float16* __restrict__ Wt,
                                                _Float16* __restrict__ Q,
                                                _Float16* __restrict__ Kb,
                                                _Float16* __restrict__ Vt) {
  const int K = DIM;
  const int bm = blockIdx.x, bn = blockIdx.y;
  const int tid = threadIdx.x;
  const int wave = tid >> 5, lane = tid & 31;
  const int wm = wave >> 1, wn = wave & 1;   // 4x2 wave grid; wave tile 32x64
  v8f acc[2][4] = {};

#if USE_TDM
  __shared__ _Float16 As[2][BM * LDP];
  __shared__ _Float16 Bs[2][BN * LDP];
  const _Float16* Abase = X  + (size_t)(bm * BM) * K;
  const _Float16* Bbase = Wt + (size_t)(bn * BN) * K;
  if (wave == 0) {
    tdm_load_tile_f16(Abase, lds_off_of(As[0]), MTOT);
    tdm_load_tile_f16(Bbase, lds_off_of(Bs[0]), 3 * DIM);
  }
  for (int kt = 0; kt < K; kt += BK) {
    const int cur = (kt >> 5) & 1;
    if (wave == 0) __builtin_amdgcn_s_wait_tensorcnt(0);
    __syncthreads();                       // tile[cur] ready; all waves done with tile[cur^1]
    if (wave == 0 && kt + BK < K) {        // DMA next tile, overlapped with WMMA below
      tdm_load_tile_f16(Abase + kt + BK, lds_off_of(As[cur ^ 1]), MTOT);
      tdm_load_tile_f16(Bbase + kt + BK, lds_off_of(Bs[cur ^ 1]), 3 * DIM);
    }
    gemm_tile_step(As[cur], Bs[cur], wm, wn, acc);
  }
#else
  __shared__ _Float16 As[BM * LDP];
  __shared__ _Float16 Bs[BN * LDP];
  const int r0 = tid >> 2, seg = (tid & 3) << 3;
  for (int kt = 0; kt < K; kt += BK) {
#pragma unroll
    for (int it = 0; it < 2; ++it) {
      int r = r0 + it * 64;
      *(uint4*)(As + r * LDP + seg) = *(const uint4*)(X  + (size_t)(bm * BM + r) * K + kt + seg);
      *(uint4*)(Bs + r * LDP + seg) = *(const uint4*)(Wt + (size_t)(bn * BN + r) * K + kt + seg);
    }
    if (kt + BK < K) {
      __builtin_prefetch(X  + (size_t)(bm * BM + r0) * K + kt + BK, 0, 1);
      __builtin_prefetch(Wt + (size_t)(bn * BN + r0) * K + kt + BK, 0, 1);
    }
    __syncthreads();
    gemm_tile_step(As, Bs, wm, wn, acc);
    __syncthreads();
  }
#endif

#pragma unroll
  for (int mi = 0; mi < 2; ++mi) {
#pragma unroll
    for (int ni = 0; ni < 4; ++ni) {
#pragma unroll
      for (int e = 0; e < 8; ++e) {
        int row = bm * BM + wm * 32 + mi * 16 + ((lane >> 4) << 3) + e;
        int col = bn * BN + wn * 64 + ni * 16 + (lane & 15);
        int b = row >> 11, t = row & (SEQ - 1);
        int which = col >> 10, c = col & (DIM - 1);
        int h = c >> 6, d = c & (HDIM - 1);
        size_t bh = (size_t)(b * NHEAD + h);
        _Float16 v = (_Float16)acc[mi][ni][e];
        if (which == 0)      Q [(bh * SEQ + t) * HDIM + d] = v;
        else if (which == 1) Kb[(bh * SEQ + t) * HDIM + d] = v;
        else                 Vt[(bh * HDIM + d) * SEQ + t] = v;
      }
    }
  }
}

// ---------------------------------------------------------------------------
// Flash attention (causal). Block = 8 waves; each wave owns 16 query rows of
// a 128-row q tile for one (b,h). Per-wave LDS staging for P (no barriers
// needed: DS ops are in-order within a wave).
// ---------------------------------------------------------------------------
__global__ __launch_bounds__(256) void attn_kernel(const _Float16* __restrict__ Q,
                                                   const _Float16* __restrict__ Kb,
                                                   const _Float16* __restrict__ Vt,
                                                   _Float16* __restrict__ O) {
  __shared__ _Float16 P[8][16 * LDP];
  const int b = blockIdx.z, h = blockIdx.y, qt = blockIdx.x;
  const int wave = threadIdx.x >> 5, lane = threadIdx.x & 31;
  const size_t bh = (size_t)(b * NHEAD + h);
  const _Float16* qp = Q  + bh * SEQ * HDIM;
  const _Float16* kp = Kb + bh * SEQ * HDIM;
  const _Float16* vp = Vt + bh * HDIM * SEQ;   // [d][T]
  const int qRow = qt * 128 + wave * 16;

  v16h qf[2];
#pragma unroll
  for (int s = 0; s < 2; ++s) qf[s] = load_a_frag(qp, qRow, s * 32, HDIM);

  float mrow[8], lrow[8];
#pragma unroll
  for (int r = 0; r < 8; ++r) { mrow[r] = -__builtin_inff(); lrow[r] = 0.f; }
  v8f oacc[4] = {};

  const int rowBase = qRow + ((lane >> 4) << 3);
  const int kEnd = qRow + 16;                  // causal: stop at the diagonal tile
  for (int k0 = 0; k0 < kEnd; k0 += 32) {
    // S = Q * K^T over d=64 (two 32-deep WMMA steps), 32 key columns
    v8f s0 = {}, s1 = {};
#pragma unroll
    for (int s = 0; s < 2; ++s) {
      v16h b0 = load_b_frag(kp, k0,      s * 32, HDIM);
      v16h b1 = load_b_frag(kp, k0 + 16, s * 32, HDIM);
      s0 = wmma_f16(qf[s], b0, s0);
      s1 = wmma_f16(qf[s], b1, s1);
    }
    const int c0 = k0 + (lane & 15);
    const int c1 = c0 + 16;
    float alpha[8];
#pragma unroll
    for (int r = 0; r < 8; ++r) {
      const int row = rowBase + r;
      float a0 = (c0 <= row) ? s0[r] * ATT_SCALE : -__builtin_inff();
      float a1 = (c1 <= row) ? s1[r] * ATT_SCALE : -__builtin_inff();
      float mx = fmaxf(a0, a1);
      mx = fmaxf(mx, __shfl_xor(mx, 1, 32));
      mx = fmaxf(mx, __shfl_xor(mx, 2, 32));
      mx = fmaxf(mx, __shfl_xor(mx, 4, 32));
      mx = fmaxf(mx, __shfl_xor(mx, 8, 32));
      float mnew = fmaxf(mrow[r], mx);
      float al = __expf(mrow[r] - mnew);
      mrow[r] = mnew;
      float p0 = __expf(a0 - mnew);
      float p1 = __expf(a1 - mnew);
      float sm = p0 + p1;
      sm += __shfl_xor(sm, 1, 32);
      sm += __shfl_xor(sm, 2, 32);
      sm += __shfl_xor(sm, 4, 32);
      sm += __shfl_xor(sm, 8, 32);
      lrow[r] = lrow[r] * al + sm;
      alpha[r] = al;
      // C-layout -> LDS so we can reload P in A-layout
      P[wave][(((lane >> 4) << 3) + r) * LDP + (lane & 15)]      = (_Float16)p0;
      P[wave][(((lane >> 4) << 3) + r) * LDP + (lane & 15) + 16] = (_Float16)p1;
    }
    v16h pa = load_a_frag(&P[wave][0], 0, 0, LDP);
    // O = O*alpha + P @ V  (V^T layout makes B frags contiguous)
#pragma unroll
    for (int ni = 0; ni < 4; ++ni) {
#pragma unroll
      for (int r = 0; r < 8; ++r) oacc[ni][r] *= alpha[r];
      v16h bv = load_b_frag(vp, ni * 16, k0, SEQ);
      oacc[ni] = wmma_f16(pa, bv, oacc[ni]);
    }
  }

  // normalize and store token-major f16 [B*T][DIM] with heads concatenated
#pragma unroll
  for (int ni = 0; ni < 4; ++ni) {
#pragma unroll
    for (int r = 0; r < 8; ++r) {
      int t = rowBase + r;
      int col = h * HDIM + ni * 16 + (lane & 15);
      O[((size_t)b * SEQ + t) * DIM + col] = (_Float16)(oacc[ni][r] / lrow[r]);
    }
  }
}

// ---------------------------------------------------------------------------
// GEMM2: out = attn16 @ w_proj + b  (Wt pre-transposed [1024][1024]), f32 out
// ---------------------------------------------------------------------------
__global__ __launch_bounds__(256) void gemm_proj(const _Float16* __restrict__ A,
                                                 const _Float16* __restrict__ Wt,
                                                 const float* __restrict__ bias,
                                                 float* __restrict__ out) {
  const int K = DIM;
  const int bm = blockIdx.x, bn = blockIdx.y;
  const int tid = threadIdx.x;
  const int wave = tid >> 5, lane = tid & 31;
  const int wm = wave >> 1, wn = wave & 1;
  v8f acc[2][4] = {};

#if USE_TDM
  __shared__ _Float16 As[2][BM * LDP];
  __shared__ _Float16 Bs[2][BN * LDP];
  const _Float16* Abase = A  + (size_t)(bm * BM) * K;
  const _Float16* Bbase = Wt + (size_t)(bn * BN) * K;
  if (wave == 0) {
    tdm_load_tile_f16(Abase, lds_off_of(As[0]), MTOT);
    tdm_load_tile_f16(Bbase, lds_off_of(Bs[0]), DIM);
  }
  for (int kt = 0; kt < K; kt += BK) {
    const int cur = (kt >> 5) & 1;
    if (wave == 0) __builtin_amdgcn_s_wait_tensorcnt(0);
    __syncthreads();
    if (wave == 0 && kt + BK < K) {
      tdm_load_tile_f16(Abase + kt + BK, lds_off_of(As[cur ^ 1]), MTOT);
      tdm_load_tile_f16(Bbase + kt + BK, lds_off_of(Bs[cur ^ 1]), DIM);
    }
    gemm_tile_step(As[cur], Bs[cur], wm, wn, acc);
  }
#else
  __shared__ _Float16 As[BM * LDP];
  __shared__ _Float16 Bs[BN * LDP];
  const int r0 = tid >> 2, seg = (tid & 3) << 3;
  for (int kt = 0; kt < K; kt += BK) {
#pragma unroll
    for (int it = 0; it < 2; ++it) {
      int r = r0 + it * 64;
      *(uint4*)(As + r * LDP + seg) = *(const uint4*)(A  + (size_t)(bm * BM + r) * K + kt + seg);
      *(uint4*)(Bs + r * LDP + seg) = *(const uint4*)(Wt + (size_t)(bn * BN + r) * K + kt + seg);
    }
    if (kt + BK < K) {
      __builtin_prefetch(A  + (size_t)(bm * BM + r0) * K + kt + BK, 0, 1);
      __builtin_prefetch(Wt + (size_t)(bn * BN + r0) * K + kt + BK, 0, 1);
    }
    __syncthreads();
    gemm_tile_step(As, Bs, wm, wn, acc);
    __syncthreads();
  }
#endif

#pragma unroll
  for (int mi = 0; mi < 2; ++mi) {
#pragma unroll
    for (int ni = 0; ni < 4; ++ni) {
#pragma unroll
      for (int e = 0; e < 8; ++e) {
        int row = bm * BM + wm * 32 + mi * 16 + ((lane >> 4) << 3) + e;
        int col = bn * BN + wn * 64 + ni * 16 + (lane & 15);
        out[(size_t)row * DIM + col] = acc[mi][ni][e] + bias[col];
      }
    }
  }
}

// ---------------------------------------------------------------------------
// Launcher
// ---------------------------------------------------------------------------
extern "C" void kernel_launch(void* const* d_in, const int* in_sizes, int n_in,
                              void* d_out, int out_size, void* d_ws, size_t ws_size,
                              hipStream_t stream) {
  (void)in_sizes; (void)n_in; (void)out_size; (void)ws_size;
  const float* x      = (const float*)d_in[0];
  // d_in[1] = key_padding_mask (all false in this problem) — unused
  const float* w_qkv  = (const float*)d_in[2];
  const float* w_proj = (const float*)d_in[3];
  const float* b_proj = (const float*)d_in[4];
  float* out = (float*)d_out;

  char* ws = (char*)d_ws;
  size_t off = 0;
  auto alloc = [&](size_t bytes) -> void* {
    void* p = ws + off;
    off = (off + bytes + 255) & ~(size_t)255;
    return p;
  };
  _Float16* x16    = (_Float16*)alloc((size_t)MTOT * DIM * 2);
  _Float16* wqkvT  = (_Float16*)alloc((size_t)3 * DIM * DIM * 2);
  _Float16* wprojT = (_Float16*)alloc((size_t)DIM * DIM * 2);
  _Float16* Qb     = (_Float16*)alloc((size_t)MTOT * DIM * 2);
  _Float16* Kb     = (_Float16*)alloc((size_t)MTOT * DIM * 2);
  _Float16* Vt     = (_Float16*)alloc((size_t)MTOT * DIM * 2);
  _Float16* Ao     = (_Float16*)alloc((size_t)MTOT * DIM * 2);

  const int nX = MTOT * DIM;
  cvt_f32_f16  <<<(nX + 255) / 256, 256, 0, stream>>>(x, x16, nX);
  cvt_transpose<<<(3 * DIM * DIM + 255) / 256, 256, 0, stream>>>(w_qkv, wqkvT, DIM, 3 * DIM);
  cvt_transpose<<<(DIM * DIM + 255) / 256, 256, 0, stream>>>(w_proj, wprojT, DIM, DIM);

  gemm_qkv   <<<dim3(MTOT / BM, 3 * DIM / BN), 256, 0, stream>>>(x16, wqkvT, Qb, Kb, Vt);
  attn_kernel<<<dim3(SEQ / 128, NHEAD, BATCH), 256, 0, stream>>>(Qb, Kb, Vt, Ao);
  gemm_proj  <<<dim3(MTOT / BM, DIM / BN),     256, 0, stream>>>(Ao, wprojT, b_proj, out);
}
